// Mamba_39444979647122
// MI455X (gfx1250) — matrix operand order repaired and checked by reading, not compiled, not run on hardware
//
#include <hip/hip_runtime.h>

// Problem constants (from reference): BATCH=128, SEQ=1024, INPUT_DIM=256, HIDDEN=512
constexpr int BB = 128;
constexpr int SS = 1024;
constexpr int DD = 256;
constexpr int HH = 512;

typedef __attribute__((ext_vector_type(16))) __bf16 v16bf;
typedef __attribute__((ext_vector_type(8)))  float  v8f;

struct U32x8 { uint4 a; uint4 b; };

static __device__ __forceinline__ unsigned short f2bf(float f) {
    union { float f; unsigned u; } v; v.f = f;
    unsigned r = v.u + 0x7FFFu + ((v.u >> 16) & 1u);   // round-to-nearest-even
    return (unsigned short)(r >> 16);
}
static __device__ __forceinline__ float bf2f(unsigned short h) {
    union { unsigned u; float f; } v; v.u = ((unsigned)h) << 16;
    return v.f;
}

// A-fragment (16x32 bf16): lane m=L&15; one contiguous 16-elt chunk at K=16*(L>>4).
static __device__ __forceinline__ v16bf ldfragA(const unsigned short* p) {
    U32x8 t; t.a = *(const uint4*)p; t.b = *(const uint4*)(p + 8);
    return __builtin_bit_cast(v16bf, t);
}
// B-fragment (32x16 bf16): lane n=L&15; one contiguous 16-elt chunk at K=16*(L>>4).
static __device__ __forceinline__ v16bf ldfragB(const unsigned short* p) {
    U32x8 t; t.a = *(const uint4*)p; t.b = *(const uint4*)(p + 8);
    return __builtin_bit_cast(v16bf, t);
}

// CDNA5 async copy: LDS[ldsoff+off] = MEM[saddr + goff + off], 16B, ASYNCcnt-tracked.
static __device__ __forceinline__ void async_b128(unsigned ldsoff, unsigned goff,
                                                  unsigned long long saddr) {
    asm volatile("global_load_async_to_lds_b128 %0, %1, %2"
                 :: "v"(ldsoff), "v"(goff), "s"(saddr) : "memory");
}
static __device__ __forceinline__ void wait_asynccnt0() {
    asm volatile("s_wait_asynccnt 0" ::: "memory");
}

// ---------------------------------------------------------------------------
// Kernel 1: W_su (768x512 f32) -> WzT bf16 [n=512][k=512], WxT bf16 [n=512][k=256]
// ---------------------------------------------------------------------------
__global__ void mamba_prep_weights(const float* __restrict__ W_su,
                                   unsigned short* __restrict__ WzT,
                                   unsigned short* __restrict__ WxT) {
    int idx = blockIdx.x * blockDim.x + threadIdx.x;
    if (idx < HH * HH) {
        int n = idx >> 9, k = idx & (HH - 1);
        WzT[idx] = f2bf(W_su[k * HH + n]);
    } else {
        int j = idx - HH * HH;
        if (j < HH * DD) {
            int n = j >> 8, k = j & (DD - 1);
            WxT[j] = f2bf(W_su[(HH + k) * HH + n]);
        }
    }
}

// ---------------------------------------------------------------------------
// Kernel 2: xw[t][b][n] (bf16) = x @ W_x + b_su   (bandwidth-bound)
// Block 256 thr (8 waves), tile M=64 x N=128; B-chunk staged via async-to-LDS.
// ---------------------------------------------------------------------------
__global__ __launch_bounds__(256) void mamba_xw_gemm(
        const float* __restrict__ x, const float* __restrict__ b_su,
        const unsigned short* __restrict__ WxT, unsigned short* __restrict__ xw) {
    __shared__ __align__(16) unsigned short lA[64 * 32];   // 4KB
    __shared__ __align__(16) unsigned short lB[128 * 32];  // 8KB
    const int tid  = threadIdx.x;
    const int wave = tid >> 5, lane = tid & 31;
    const int lo = lane & 15, hl = lane >> 4;
    const int mw = wave & 3, nh = wave >> 2;
    const int m0 = blockIdx.x * 64;
    const int n0 = blockIdx.y * 128;
    const unsigned long long wxb = (unsigned long long)WxT;

    v8f acc[4] = {};

    for (int kk = 0; kk < DD; kk += 32) {
        {   // stage B async: 128 rows x 16 k bf16 (k = c..c+7, c = 0/16); 16B per thread via one b128
            int r = tid >> 1, c = (tid & 1) * 16;
            unsigned ldsoff = (unsigned)(size_t)&lB[r * 32 + c];
            unsigned goff   = (unsigned)(((n0 + r) * DD + kk + c) * 2);
            async_b128(ldsoff, goff, wxb);
        }
        {   // stage A through VALU (f32 -> bf16): 4 threads/row, 8 floats each
            int r = tid >> 2, c = (tid & 3) * 8;
            const float* src = x + (size_t)(m0 + r) * DD + kk + c;
            float4 f0 = *(const float4*)src;
            float4 f1 = *(const float4*)(src + 4);
            unsigned short* dst = &lA[r * 32 + c];
            dst[0] = f2bf(f0.x); dst[1] = f2bf(f0.y); dst[2] = f2bf(f0.z); dst[3] = f2bf(f0.w);
            dst[4] = f2bf(f1.x); dst[5] = f2bf(f1.y); dst[6] = f2bf(f1.z); dst[7] = f2bf(f1.w);
        }
        wait_asynccnt0();
        __syncthreads();
        v16bf a = ldfragA(&lA[(mw * 16 + lo) * 32 + 16 * hl]);
        #pragma unroll
        for (int j = 0; j < 4; ++j) {
            v16bf b = ldfragB(&lB[(nh * 64 + j * 16 + lo) * 32 + 16 * hl]);
            acc[j] = __builtin_amdgcn_wmma_f32_16x16x32_bf16(
                         false, a, false, b, (short)0, acc[j], false, false);
        }
        __syncthreads();
    }
    #pragma unroll
    for (int j = 0; j < 4; ++j) {
        int n = n0 + nh * 64 + j * 16 + lo;
        float bias = b_su[n];
        #pragma unroll
        for (int r = 0; r < 8; ++r) {
            int row = m0 + mw * 16 + r + 8 * hl;       // row = b*SEQ + t
            int b = row >> 10, t = row & (SS - 1);
            xw[((size_t)t * BB + b) * HH + n] = f2bf(acc[j][r] + bias);
        }
    }
}

// ---------------------------------------------------------------------------
// Kernel 3: sequential scan. 8 workgroups x 16 batch rows; z double-buffered in
// LDS; 32 waves, one 16-col N-tile each. Half of each wave's W_z strip is
// pinned in VGPRs (64 VGPRs), half streamed from L2. xW step slices staged
// one step ahead via async-to-LDS (ASYNCcnt). One s_barrier per step.
// ---------------------------------------------------------------------------
__global__ __launch_bounds__(1024) void mamba_scan(
        const float* __restrict__ carry,
        const unsigned short* __restrict__ xw,
        const unsigned short* __restrict__ WzT,
        const float* __restrict__ W_out, const float* __restrict__ b_out,
        float* __restrict__ out) {
    __shared__ __align__(16) unsigned short zb[2][16 * HH];    // 32KB
    __shared__ __align__(16) unsigned short xwb[2][16 * HH];   // 32KB
    const int tid  = threadIdx.x;
    const int wave = tid >> 5, lane = tid & 31;
    const int lo = lane & 15, hl = lane >> 4;
    const int n0w = wave * 16;
    const int b0  = blockIdx.x * 16;
    const unsigned long long xwbase = (unsigned long long)xw;

    // pin first-half K (kt=0..7) B-fragments of this wave's W strip in VGPRs
    v16bf Bpin[8];
    #pragma unroll
    for (int kt = 0; kt < 8; ++kt)
        Bpin[kt] = ldfragB(&WzT[(size_t)(n0w + lo) * HH + kt * 32 + 16 * hl]);
    // warm L2/L0 for the streamed half
    __builtin_prefetch(&WzT[(size_t)(n0w + lo) * HH + 256 + 16 * hl], 0, 2);

    {   // init z0 = carry (16x512), 8 contiguous elements per thread
        int e = tid * 8;
        int m = e >> 9, n = e & (HH - 1);
        const float* src = carry + (size_t)(b0 + m) * HH + n;
        float4 f0 = *(const float4*)src;
        float4 f1 = *(const float4*)(src + 4);
        unsigned short* dst = &zb[0][e];
        dst[0] = f2bf(f0.x); dst[1] = f2bf(f0.y); dst[2] = f2bf(f0.z); dst[3] = f2bf(f0.w);
        dst[4] = f2bf(f1.x); dst[5] = f2bf(f1.y); dst[6] = f2bf(f1.z); dst[7] = f2bf(f1.w);
    }
    // async-stage xW slice for t=0
    async_b128((unsigned)(size_t)&xwb[0][tid * 8],
               (unsigned)((((size_t)0 * BB + b0) * HH + tid * 8) * 2), xwbase);
    wait_asynccnt0();
    __syncthreads();

    const float TWO_LOG2E = 2.8853900817779268f;   // 2/ln(2)
    for (int t = 0; t < SS; ++t) {
        const unsigned short* cur = zb[t & 1];
        unsigned short*       nxt = zb[(t + 1) & 1];
        const unsigned short* xws = xwb[t & 1];

        // kick async copy of next step's xW slice (overlaps this step's WMMAs)
        if (t + 1 < SS)
            async_b128((unsigned)(size_t)&xwb[(t + 1) & 1][tid * 8],
                       (unsigned)((((size_t)(t + 1) * BB + b0) * HH + tid * 8) * 2),
                       xwbase);

        // acc <- xW_t tile from LDS (bias folded in at phase 1)
        v8f acc;
        #pragma unroll
        for (int r = 0; r < 8; ++r)
            acc[r] = bf2f(xws[(r + 8 * hl) * HH + n0w + lo]);

        // K = 0..255: register-pinned B
        #pragma unroll
        for (int kt = 0; kt < 8; ++kt) {
            v16bf a = ldfragA(&cur[lo * HH + kt * 32 + 16 * hl]);
            acc = __builtin_amdgcn_wmma_f32_16x16x32_bf16(
                      false, a, false, Bpin[kt], (short)0, acc, false, false);
        }
        // K = 256..511: streamed from L2 (pointer laundered so loads stay in-loop
        // instead of being hoisted into 64 more live VGPRs)
        const unsigned short* wz = WzT;
        asm volatile("" : "+s"(wz));
        #pragma unroll
        for (int kt = 8; kt < 16; ++kt) {
            v16bf a = ldfragA(&cur[lo * HH + kt * 32 + 16 * hl]);
            v16bf b = ldfragB(&wz[(size_t)(n0w + lo) * HH + kt * 32 + 16 * hl]);
            acc = __builtin_amdgcn_wmma_f32_16x16x32_bf16(
                      false, a, false, b, (short)0, acc, false, false);
        }

        #pragma unroll
        for (int r = 0; r < 8; ++r) {
            int m = r + 8 * hl;
            float e = __builtin_amdgcn_exp2f(acc[r] * TWO_LOG2E);  // e^(2x)
            float z = 1.0f - 2.0f / (e + 1.0f);                    // tanh, inf-safe
            nxt[m * HH + n0w + lo] = f2bf(z);
            if (t == SS - 1)                                       // z_last, f32
                out[BB + (size_t)(b0 + m) * HH + n0w + lo] = z;
        }
        wait_asynccnt0();   // our slice-copy for t+1 has landed
        __syncthreads();    // everyone's has; z/xW buffers consistent
    }

    // out = z_last @ W_out + b_out  (z_last sits in zb[0] since SEQ is even)
    if (wave < 16) {
        const unsigned short* zl = &zb[0][wave * HH];
        float p = 0.0f;
        for (int i = lane; i < HH; i += 32) p += bf2f(zl[i]) * W_out[i];
        #pragma unroll
        for (int off = 16; off > 0; off >>= 1) p += __shfl_down(p, off, 32);
        if (lane == 0) out[b0 + wave] = p + b_out[0];
    }
}

// ---------------------------------------------------------------------------
extern "C" void kernel_launch(void* const* d_in, const int* in_sizes, int n_in,
                              void* d_out, int out_size, void* d_ws, size_t ws_size,
                              hipStream_t stream) {
    const float* x     = (const float*)d_in[0];
    const float* carry = (const float*)d_in[1];
    const float* W_su  = (const float*)d_in[2];
    const float* b_su  = (const float*)d_in[3];
    const float* W_out = (const float*)d_in[4];
    const float* b_out = (const float*)d_in[5];
    float* out = (float*)d_out;

    // workspace (bf16): xw [SEQ][BATCH][HIDDEN], WzT [512][512], WxT [512][256]
    unsigned short* xw  = (unsigned short*)d_ws;
    unsigned short* WzT = xw  + (size_t)SS * BB * HH;
    unsigned short* WxT = WzT + (size_t)HH * HH;

    mamba_prep_weights<<<dim3((HH * HH + HH * DD) / 256), dim3(256), 0, stream>>>(W_su, WzT, WxT);
    mamba_xw_gemm<<<dim3((BB * SS) / 64, HH / 128), dim3(256), 0, stream>>>(x, b_su, WxT, xw);
    mamba_scan<<<dim3(BB / 16), dim3(1024), 0, stream>>>(carry, xw, WzT, W_out, b_out, out);
}
